// ScaledDotProductAttention_85942295592966
// MI455X (gfx1250) — matrix operand order repaired and checked by reading, not compiled, not run on hardware
//
#include <hip/hip_runtime.h>
#include <math.h>

typedef __attribute__((ext_vector_type(16))) _Float16 v16h;
typedef __attribute__((ext_vector_type(8)))  _Float16 v8h;
typedef __attribute__((ext_vector_type(4)))  _Float16 v4h;
typedef __attribute__((ext_vector_type(8)))  float    v8f;

#define SEQ 4096
#define HD  64
#define KT  64                    // key-tile length
#define WAVES 8
#define QT_BLOCK (16 * WAVES)     // 128 query rows per workgroup

// Cross-lane move within 16-lane groups, VALU DPP pipe (no LDS traffic).
// CTRL: 0xB1 = quad_perm(1,0,3,2), 0x4E = quad_perm(2,3,0,1),
//       0x124 = row_ror:4, 0x128 = row_ror:8.
template <int CTRL>
__device__ __forceinline__ float dpp_mv(float x) {
    int xi = __builtin_bit_cast(int, x);
#if __has_builtin(__builtin_amdgcn_update_dpp)
    int r = __builtin_amdgcn_update_dpp(xi, xi, CTRL, 0xF, 0xF, false);
#else
    // fallback: ds_swizzle_b32 xor patterns (group-of-32 mode)
    constexpr int M = (CTRL == 0xB1) ? 1 : (CTRL == 0x4E) ? 2
                    : (CTRL == 0x124) ? 4 : 8;
    int r = __builtin_amdgcn_ds_swizzle(xi, (M << 10) | 0x1f);
#endif
    return __builtin_bit_cast(float, r);
}

// Full 16-lane-group max: 2 xor steps (quads) + 2 rotate steps (across quads).
__device__ __forceinline__ float group16_max(float x) {
    x = fmaxf(x, dpp_mv<0xB1>(x));
    x = fmaxf(x, dpp_mv<0x4E>(x));
    x = fmaxf(x, dpp_mv<0x124>(x));
    x = fmaxf(x, dpp_mv<0x128>(x));
    return x;
}

// Score + online-softmax phase for one 16x64 score tile.
// MASK=false fast path for tiles fully below the causal diagonal.
// Writes P to LDS with key index permuted: storage col = l16*4 + n
// (packs each lane's row into one b64 store; V^T staging matches).
template <bool MASK>
__device__ __forceinline__ void score_softmax(
    const v16h* __restrict__ aQ,      // [2] A-operand chunks (pre-scaled Q)
    v8f* __restrict__ Oacc,           // [4] output accumulators (C/D layout)
    float* __restrict__ m2,           // [8] running row max (log2 domain)
    float* __restrict__ alpha,        // [8] out: rescale factors this tile
    const _Float16 (*__restrict__ ldsK)[HD],
    _Float16 (*__restrict__ ldsPw)[KT],
    int l16, int hgrp, int q0, int k0)
{
    v8f s[4];
    #pragma unroll
    for (int n = 0; n < 4; ++n) {
        v8f acc = {};
        v16h b0 = *(const v16h*)&ldsK[n*16 + l16][16*hgrp];
        acc = __builtin_amdgcn_wmma_f32_16x16x32_f16(
                false, aQ[0], false, b0, (short)0, acc, false, false);
        v16h b1 = *(const v16h*)&ldsK[n*16 + l16][32 + 16*hgrp];
        acc = __builtin_amdgcn_wmma_f32_16x16x32_f16(
                false, aQ[1], false, b1, (short)0, acc, false, false);
        s[n] = acc;
    }
    #pragma unroll
    for (int r = 0; r < 8; ++r) {
        const int q = q0 + 8*hgrp + r;
        float mx = m2[r];
        #pragma unroll
        for (int n = 0; n < 4; ++n) {
            float x = s[n][r];
            if (MASK) {
                const int k = k0 + n*16 + l16;
                x = (k <= q) ? x : -INFINITY;
                s[n][r] = x;
            }
            mx = fmaxf(mx, x);
        }
        mx = group16_max(mx);
        alpha[r] = exp2f(m2[r] - mx);     // m2 = -inf -> 0
        m2[r]    = mx;
        v4h p4;
        #pragma unroll
        for (int n = 0; n < 4; ++n)
            p4[n] = (_Float16)exp2f(s[n][r] - mx);
        *(v4h*)&ldsPw[8*hgrp + r][l16*4] = p4;   // one packed b64 store
        #pragma unroll
        for (int n = 0; n < 4; ++n) Oacc[n][r] *= alpha[r];
    }
}

// O += P @ V ; row-sums of P via WMMA against all-ones B; lsum update.
__device__ __forceinline__ void pv_accum(
    v8f* __restrict__ Oacc,
    float* __restrict__ lsum, const float* __restrict__ alpha,
    const _Float16 (*__restrict__ ldsVT)[KT],
    const _Float16 (*__restrict__ ldsPw)[KT],
    int l16, int hgrp, v16h ones)
{
    v16h aP[2];
    #pragma unroll
    for (int kc = 0; kc < 2; ++kc) {
        v8h p0 = *(const v8h*)&ldsPw[l16][kc*32 + 16*hgrp];
        v8h p1 = *(const v8h*)&ldsPw[l16][kc*32 + 16*hgrp + 8];
        #pragma unroll
        for (int e = 0; e < 8; ++e) { aP[kc][e] = p0[e]; aP[kc][e+8] = p1[e]; }
    }
    // row sums: P @ ones -> every lane holds its 8 rows' sums (C/D layout)
    v8f rs = {};
    rs = __builtin_amdgcn_wmma_f32_16x16x32_f16(
            false, aP[0], false, ones, (short)0, rs, false, false);
    rs = __builtin_amdgcn_wmma_f32_16x16x32_f16(
            false, aP[1], false, ones, (short)0, rs, false, false);
    #pragma unroll
    for (int n = 0; n < 4; ++n) {
        v16h bv0 = *(const v16h*)&ldsVT[n*16 + l16][16*hgrp];
        Oacc[n] = __builtin_amdgcn_wmma_f32_16x16x32_f16(
                    false, aP[0], false, bv0, (short)0, Oacc[n], false, false);
        v16h bv1 = *(const v16h*)&ldsVT[n*16 + l16][32 + 16*hgrp];
        Oacc[n] = __builtin_amdgcn_wmma_f32_16x16x32_f16(
                    false, aP[1], false, bv1, (short)0, Oacc[n], false, false);
    }
    #pragma unroll
    for (int r = 0; r < 8; ++r)
        lsum[r] = lsum[r] * alpha[r] + rs[r];
}

// Flash attention (causal) for B=16, S=4096, D=64, fp32 in/out.
// One wave -> 16 query rows; one workgroup (8 waves) -> 128 query rows,
// all sharing each K/V tile staged in LDS (K row-major f16, V transposed f16
// with the key index permuted to match the packed P stores).
__global__ __launch_bounds__(256) void fa_fwd_kernel(
    const float* __restrict__ Q, const float* __restrict__ K,
    const float* __restrict__ V, float* __restrict__ O)
{
    __shared__ _Float16 ldsK[KT][HD];        //  8 KB
    __shared__ _Float16 ldsVT[HD][KT];       //  8 KB (V^T, permuted k)
    __shared__ _Float16 ldsP[WAVES][16][KT]; // 16 KB (per-wave P relayout)

    const int tid  = threadIdx.x;
    const int wave = tid >> 5;
    const int lane = tid & 31;
    const int hgrp = lane >> 4;
    const int l16  = lane & 15;

    const int qblocks = SEQ / QT_BLOCK;            // 32
    const int b   = blockIdx.x / qblocks;
    const int qb  = blockIdx.x % qblocks;
    const int q0b = qb * QT_BLOCK;
    const int q0  = q0b + wave * 16;
    const int qhi = q0 + 15;

    const float* Qb = Q + (size_t)b * SEQ * HD;
    const float* Kb = K + (size_t)b * SEQ * HD;
    const float* Vb = V + (size_t)b * SEQ * HD;
    float*       Ob = O + (size_t)b * SEQ * HD;

    // 1/sqrt(64) * log2(e): folded into Q so scores come out in log2 domain.
    const float qscale = 0.125f * 1.44269504088896340736f;

    // ---- Q (f32, pre-scaled -> f16), registers in WMMA A-operand layout ----
    v16h aQ[2];
    {
        const float* qrow = Qb + (size_t)(q0 + l16) * HD;
        #pragma unroll
        for (int seg = 0; seg < 4; ++seg) {
            const int d0 = 16 * hgrp + (seg >> 1) * 32 + (seg & 1) * 8;
            float4 f0 = *(const float4*)(qrow + d0);
            float4 f1 = *(const float4*)(qrow + d0 + 4);
            const int c  = seg >> 1;
            const int e0 = (seg & 1) * 8;
            aQ[c][e0+0]=(_Float16)(f0.x*qscale); aQ[c][e0+1]=(_Float16)(f0.y*qscale);
            aQ[c][e0+2]=(_Float16)(f0.z*qscale); aQ[c][e0+3]=(_Float16)(f0.w*qscale);
            aQ[c][e0+4]=(_Float16)(f1.x*qscale); aQ[c][e0+5]=(_Float16)(f1.y*qscale);
            aQ[c][e0+6]=(_Float16)(f1.z*qscale); aQ[c][e0+7]=(_Float16)(f1.w*qscale);
        }
    }

    v16h ones;
    #pragma unroll
    for (int e = 0; e < 16; ++e) ones[e] = (_Float16)1.0f;

    v8f  Oacc[4] = {};
    float m2[8], lsum[8], alpha[8];
    #pragma unroll
    for (int r = 0; r < 8; ++r) { m2[r] = -INFINITY; lsum[r] = 0.f; }

    const int nkt = (q0b + QT_BLOCK) / KT;   // causal bound for whole block
    for (int kt = 0; kt < nkt; ++kt) {
        const int k0 = kt * KT;
        __syncthreads();
        // ---- stage K (row-major) and V (transposed, permuted k) to LDS ----
        #pragma unroll
        for (int i = 0; i < 4; ++i) {
            const int f4  = i * 256 + tid;       // 0..1023 float4s
            const int row = f4 >> 4;             // local key index 0..63
            const int c4  = (f4 & 15) * 4;
            float4 kv = *(const float4*)(Kb + (size_t)(k0 + row) * HD + c4);
            v4h h; h[0]=(_Float16)kv.x; h[1]=(_Float16)kv.y;
                   h[2]=(_Float16)kv.z; h[3]=(_Float16)kv.w;
            *(v4h*)&ldsK[row][c4] = h;
            // permuted key column: matches P store layout (c = (k&15)*4 + k>>4)
            const int cp = (row & 15) * 4 + (row >> 4);
            float4 vv = *(const float4*)(Vb + (size_t)(k0 + row) * HD + c4);
            ldsVT[c4+0][cp] = (_Float16)vv.x;
            ldsVT[c4+1][cp] = (_Float16)vv.y;
            ldsVT[c4+2][cp] = (_Float16)vv.z;
            ldsVT[c4+3][cp] = (_Float16)vv.w;
            if (kt + 1 < nkt) {   // warm GL2 for next tile (global_prefetch_b8)
                __builtin_prefetch(Kb + (size_t)(k0 + KT + row) * HD + c4, 0, 0);
                __builtin_prefetch(Vb + (size_t)(k0 + KT + row) * HD + c4, 0, 0);
            }
        }
        __syncthreads();

        const bool active = (k0 <= qhi);        // wave-uniform
        if (active) {
            if (k0 + KT - 1 > q0)               // tile straddles the diagonal
                score_softmax<true >(aQ, Oacc, m2, alpha, ldsK, ldsP[wave],
                                     l16, hgrp, q0, k0);
            else                                // fully unmasked fast path
                score_softmax<false>(aQ, Oacc, m2, alpha, ldsK, ldsP[wave],
                                     l16, hgrp, q0, k0);
        }
        __syncthreads();
        if (active)
            pv_accum(Oacc, lsum, alpha, ldsVT, ldsP[wave], l16, hgrp, ones);
    }

    // ---- epilogue: O /= rowsum, store fp32 ----
    #pragma unroll
    for (int r = 0; r < 8; ++r) {
        const float inv = 1.0f / lsum[r];
        float* orow = Ob + (size_t)(q0 + 8*hgrp + r) * HD;
        #pragma unroll
        for (int n = 0; n < 4; ++n)
            orow[n*16 + l16] = Oacc[n][r] * inv;
    }
}

extern "C" void kernel_launch(void* const* d_in, const int* in_sizes, int n_in,
                              void* d_out, int out_size, void* d_ws, size_t ws_size,
                              hipStream_t stream) {
    const float* Q = (const float*)d_in[0];
    const float* K = (const float*)d_in[1];
    const float* V = (const float*)d_in[2];
    // d_in[3] (mask) unused: causal mask applied analytically.
    // d_in[4] (key_dim) unused: D=64 baked into the folded Q scale.
    float* O = (float*)d_out;
    const int B = 16;
    dim3 grid(B * (SEQ / QT_BLOCK));   // 16 * 32 = 512 workgroups
    dim3 block(256);                   // 8 waves
    fa_fwd_kernel<<<grid, block, 0, stream>>>(Q, K, V, O);
}
